// SSIM_300647711406
// MI455X (gfx1250) — matrix operand order repaired and checked by reading, not compiled, not run on hardware
//
#include <hip/hip_runtime.h>
#include <stdint.h>

typedef __attribute__((ext_vector_type(16))) _Float16 v16h;
typedef __attribute__((ext_vector_type(8)))  float    v8f;
typedef __attribute__((ext_vector_type(4)))  unsigned v4u;

#define IMG    512
#define TILE   64
#define HALO   5
#define LW     74            // TILE + 2*HALO
#define SROW   75            // f32 staging row stride (odd => fewer bank conflicts)
#define PPAIR  40            // padded row/col pair count (80 rows/cols of f16)
#define TPB    256
#define NBLK   (96 * 64)     // 32*3 planes, 8x8 tiles each
#define C1SSIM 1.0e-4f
#define C2SSIM 9.0e-4f

#if defined(__has_builtin)
#if __has_builtin(__builtin_amdgcn_global_load_async_to_lds_b32)
#define USE_ASYNC_LDS 1
#endif
#endif

typedef __attribute__((address_space(1))) int g_int;
typedef __attribute__((address_space(3))) int l_int;

__device__ inline void wait_async0() {
#if defined(__has_builtin) && __has_builtin(__builtin_amdgcn_s_wait_asynccnt)
  __builtin_amdgcn_s_wait_asynccnt(0);
#else
  asm volatile("s_wait_asynccnt 0" ::: "memory");
#endif
}

__device__ inline unsigned pk2h(float lo, float hi) {
  union { _Float16 h[2]; unsigned u; } z;
  z.h[0] = (_Float16)lo; z.h[1] = (_Float16)hi;
  return z.u;
}

__global__ __launch_bounds__(TPB) void ssim_tile(const float* __restrict__ g1,
                                                 const float* __restrict__ g2,
                                                 float* __restrict__ partial) {
  __shared__ float    sA[LW * SROW];
  __shared__ float    sB[LW * SROW];
  __shared__ unsigned P[5][LW * PPAIR];     // [quantity][col][row-pair]  (col-major)
  __shared__ unsigned I[5][TILE * PPAIR];   // [quantity][row][col-pair]  (row-major)
  __shared__ unsigned WT[48];               // packed weight pairs: (w(t-16), w(t-15))
  __shared__ float    red[TPB];

  const int tid  = threadIdx.x;
  const int lane = tid & 31;
  const int wave = tid >> 5;
  const int bid  = blockIdx.x;
  const int plane = bid >> 6;
  const int t     = bid & 63;
  const int ty = (t >> 3) * TILE;
  const int tx = (t & 7) * TILE;
  const int y0 = ty - HALO, x0 = tx - HALO;
  const float* p1 = g1 + (size_t)plane * IMG * IMG;
  const float* p2 = g2 + (size_t)plane * IMG * IMG;

  // ---- Phase 1: stage img tiles (zero halo) into LDS via async copies ----
  {
    int r = tid / LW, c = tid - r * LW;   // one division, then incremental
    for (int i = tid; i < LW * LW; i += TPB) {
      int gy = y0 + r, gx = x0 + c;
      int l  = r * SROW + c;
      bool ok = (gy >= 0) & (gy < IMG) & (gx >= 0) & (gx < IMG);
      if (ok) {
        size_t gi = (size_t)gy * IMG + gx;
#if USE_ASYNC_LDS
        __builtin_amdgcn_global_load_async_to_lds_b32(
            (g_int*)(p1 + gi), (l_int*)&sA[l], 0, 0);
        __builtin_amdgcn_global_load_async_to_lds_b32(
            (g_int*)(p2 + gi), (l_int*)&sB[l], 0, 0);
#else
        sA[l] = p1[gi];
        sB[l] = p2[gi];
#endif
      } else {
        sA[l] = 0.0f;
        sB[l] = 0.0f;
      }
      r += 3; c += 34;                    // += TPB over width LW (256 = 3*74+34)
      if (c >= LW) { c -= LW; ++r; }
    }
  }

  // ---- Weight pair-table (48 entries), built by 48 threads ----
  if (tid < 48) {
    float wsum = 0.0f;
#pragma unroll
    for (int j = 0; j < 11; ++j) {
      float d = (float)(j - 5);
      wsum += expf(-d * d / 4.5f);
    }
    float winv = 1.0f / wsum;
    int j0 = tid - 16;                    // lo half weight index
    float dl = (float)(j0 - 5), dh = (float)(j0 - 4);
    float wl = (j0 >= 0 && j0 <= 10) ? expf(-dl * dl / 4.5f) * winv : 0.0f;
    float wh = (j0 + 1 >= 0 && j0 + 1 <= 10) ? expf(-dh * dh / 4.5f) * winv : 0.0f;
    WT[tid] = pk2h(wl, wh);
  }

  wait_async0();
  __syncthreads();

  const int nlo  = lane & 15;
  const int hiB  = lane >> 4;           // 0/1: upper half-wave
  const int hi8  = hiB << 3;
  const int hi16 = hiB << 4;

  // Band weight fragment, shared by both passes: B[k][n] = w[k-n]  (32x16 f16)
  union { v16h v; unsigned u[8]; } bW;
#pragma unroll
  for (int v = 0; v < 8; ++v) {
    int klo = (v < 4) ? (2 * v + hi8) : (16 + hi8 + 2 * (v - 4));
    bW.u[v] = WT[klo - nlo + 16];
  }

  // ---- Phase 2: quantities -> packed f16 row pairs, column-major ----
  {
    int c = tid / PPAIR, p = tid - (tid / PPAIR) * PPAIR;
    for (int i = tid; i < LW * PPAIR; i += TPB) {
      int r0 = 2 * p;
      float xa = 0.f, xb = 0.f, ya = 0.f, yb = 0.f;
      if (r0 < LW)     { xa = sA[r0 * SROW + c];       ya = sB[r0 * SROW + c]; }
      if (r0 + 1 < LW) { xb = sA[(r0 + 1) * SROW + c]; yb = sB[(r0 + 1) * SROW + c]; }
      int o = c * PPAIR + p;
      P[0][o] = pk2h(xa, xb);
      P[1][o] = pk2h(ya, yb);
      P[2][o] = pk2h(xa * xa, xb * xb);
      P[3][o] = pk2h(ya * ya, yb * yb);
      P[4][o] = pk2h(xa * ya, xb * yb);
      c += 6; p += 16;                    // += TPB over PPAIR (256 = 6*40+16)
      if (p >= PPAIR) { p -= PPAIR; ++c; }
    }
  }
  __syncthreads();

  // ---- Phase 3: vertical blur via WMMA: D = Data(16x32) x W_band(32x16) ----
  // A rows = 16 interm columns, K = 32 input rows, D cols = 16 output rows.
  // 4 row-tiles x 5 col-tiles x 5 quantities = 100 tasks over 8 waves
  for (int task = wave; task < 100; task += 8) {
    int q = task / 20;
    int s = task - q * 20;
    int R0 = (s / 5) * 16;       // output-row base within tile (0..48)
    int C0 = (s % 5) * 16;       // interm-col base (0..64)
    union { v16h v; unsigned u[8]; } aD;
    int col = C0 + nlo;
    if (col > LW - 1) col = LW - 1;        // clamp pad cols: finite, x0 weight
    int pbase = col * PPAIR + (R0 >> 1) + (hi8 >> 1);
#pragma unroll
    for (int v = 0; v < 8; ++v) {
      int pr = (v < 4) ? v : (8 + (v - 4));   // klo/2 (minus the hi part)
      aD.u[v] = P[q][pbase + pr];
    }
    v8f z = {};
    v8f d = __builtin_amdgcn_wmma_f32_16x16x32_f16(false, aD.v, false, bW.v,
                                                   (short)0, z, false, false);
    // lane holds out-row R0+nlo, interm cols C0+hi8+0..7 -> one aligned b128
    union { v4u v; unsigned u[4]; } st;
#pragma unroll
    for (int j = 0; j < 4; ++j) st.u[j] = pk2h(d[2 * j], d[2 * j + 1]);
    *(v4u*)&I[q][(R0 + nlo) * PPAIR + (C0 >> 1) + (hi8 >> 1)] = st.v;
  }
  __syncthreads();

  // ---- Phase 4: horizontal blur via WMMA + SSIM map ----
  float accs = 0.0f;
  for (int t2 = wave; t2 < 16; t2 += 8) {
    int R0 = (t2 >> 2) * 16;
    int C0 = (t2 & 3) * 16;
    v8f D[5];
#pragma unroll
    for (int q = 0; q < 5; ++q) {
      union { v16h v; unsigned u[8]; } aD;
      int base = (R0 + nlo) * PPAIR + (C0 >> 1) + (hi8 >> 1);
#pragma unroll
      for (int v = 0; v < 8; ++v) {
        int pr = (v < 4) ? v : (8 + (v - 4));
        aD.u[v] = I[q][base + pr];
      }
      v8f z = {};
      D[q] = __builtin_amdgcn_wmma_f32_16x16x32_f16(false, aD.v, false, bW.v,
                                                    (short)0, z, false, false);
    }
#pragma unroll
    for (int r = 0; r < 8; ++r) {
      float mu1 = D[0][r], mu2 = D[1][r];
      float m11 = mu1 * mu1, m22 = mu2 * mu2, m12 = mu1 * mu2;
      float s11 = D[2][r] - m11;
      float s22 = D[3][r] - m22;
      float s12 = D[4][r] - m12;
      float num = (2.0f * m12 + C1SSIM) * (2.0f * s12 + C2SSIM);
      float den = (m11 + m22 + C1SSIM) * (s11 + s22 + C2SSIM);
      accs += num / den;
    }
  }

  // ---- deterministic in-block reduction ----
  red[tid] = accs;
  __syncthreads();
  for (int s = TPB / 2; s > 0; s >>= 1) {
    if (tid < s) red[tid] += red[tid + s];
    __syncthreads();
  }
  if (tid == 0) partial[bid] = red[0];
}

__global__ __launch_bounds__(TPB) void ssim_reduce(const float* __restrict__ partial,
                                                   float* __restrict__ out) {
  __shared__ float red[TPB];
  int tid = threadIdx.x;
  float s = 0.0f;
#pragma unroll 4
  for (int i = 0; i < NBLK / TPB; ++i) s += partial[tid * (NBLK / TPB) + i];
  red[tid] = s;
  __syncthreads();
  for (int k = TPB / 2; k > 0; k >>= 1) {
    if (tid < k) red[tid] += red[tid + k];
    __syncthreads();
  }
  if (tid == 0) out[0] = red[0] * (1.0f / 25165824.0f);  // / (32*3*512*512)
}

extern "C" void kernel_launch(void* const* d_in, const int* in_sizes, int n_in,
                              void* d_out, int out_size, void* d_ws, size_t ws_size,
                              hipStream_t stream) {
  (void)in_sizes; (void)n_in; (void)out_size; (void)ws_size;
  const float* img1 = (const float*)d_in[0];
  const float* img2 = (const float*)d_in[1];
  float* ws  = (float*)d_ws;    // NBLK partial sums (24 KB)
  float* out = (float*)d_out;

  ssim_tile<<<dim3(NBLK), dim3(TPB), 0, stream>>>(img1, img2, ws);
  ssim_reduce<<<dim3(1), dim3(TPB), 0, stream>>>(ws, out);
}
